// DiffusionPriorNetwork_86191403696372
// MI455X (gfx1250) — matrix operand order repaired and checked by reading, not compiled, not run on hardware
//
#include <hip/hip_runtime.h>
#include <math.h>

// ---------------------------------------------------------------------------
// DiffusionPriorNetwork forward for MI455X (gfx1250, wave32, WMMA).
// Compute-bound (~4.1 TFLOP vs ~60us of HBM traffic at 23.3 TB/s), so all
// large GEMMs run on v_wmma_f32_16x16x32_bf16 with fp32 accumulation.
// GEMM K-loop is software-pipelined with double-buffered LDS so global
// loads for tile k+1 overlap the WMMA chain on tile k.
// ---------------------------------------------------------------------------

#define B_     256
#define L_     77
#define DIM_   768
#define DEPTH_ 12
#define HEADS_ 12
#define DH_    64
#define NTOK   80          // 77 text + text_embed + time_embed + query
#define NKEY   81          // null kv prepended
#define INNER_ 768
#define FFIN   3072        // per half of the SwiGLU projection
#define M_     (B_*NTOK)   // 20480 rows in every big GEMM

typedef float  v8f   __attribute__((ext_vector_type(8)));
typedef __bf16 v16bf __attribute__((ext_vector_type(16)));
typedef unsigned int u32x4 __attribute__((ext_vector_type(4)));

union FragBF { v16bf v; u32x4 u[2]; };   // 32 bytes, trivial types only

__device__ __forceinline__ unsigned short f2bf(float f) {
  unsigned int u = __float_as_uint(f);
  u += 0x7FFFu + ((u >> 16) & 1u);       // round-to-nearest-even
  return (unsigned short)(u >> 16);
}
__device__ __forceinline__ float bf2f(unsigned short h) {
  return __uint_as_float(((unsigned int)h) << 16);
}

// ---------------------------------------------------------------------------
// Tiled bf16 WMMA GEMM:  C(MxN) = A(MxK, bf16 row-major) @ W, where W is
// supplied pre-transposed Bt(NxK, bf16 row-major).  Block tile 128x128x32,
// 8 waves; each wave computes a 32x64 tile with 2x4 v_wmma_f32_16x16x32_bf16.
// Double-buffered LDS + register staging: one barrier per K step, global
// loads for the next tile issue before the current tile's WMMAs.
// EPI: 0 = store f32, 1 = store bf16, 2 = fp32 residual add in place.
// M,N multiples of 128; K multiple of 32.
// ---------------------------------------------------------------------------
#define BM 128
#define BN 128
#define BK 32
#define LDSW 40   // halfword row stride in LDS (32 + 8 pad, keeps 16B align)

template<int EPI>
__global__ __launch_bounds__(256) void gemm_bf16_wmma(
    const unsigned short* __restrict__ A,
    const unsigned short* __restrict__ Bt,
    void* __restrict__ Cv,
    int M, int N, int K)
{
  __shared__ unsigned short As[2][BM * LDSW];
  __shared__ unsigned short Bs[2][BN * LDSW];

  const int m0   = blockIdx.y * BM;
  const int n0   = blockIdx.x * BN;
  const int t    = threadIdx.x;
  const int wave = t >> 5;
  const int lane = t & 31;
  const int hh   = lane >> 4;    // lane half (ISA fragment layouts)
  const int rr   = lane & 15;
  const int wm   = wave & 3;     // wave M tile: 4 waves along M (32 rows each)
  const int wn   = wave >> 2;    // wave N tile: 2 waves along N (64 cols each)
  const int lrow = t >> 1;       // 128 rows, 2 threads/row for tile loads
  const int lclh = (t & 1) << 4; // halfword offset 0 or 16

  v8f acc[2][4] = {};

  const size_t aRow = (size_t)(m0 + lrow) * K;
  const size_t bRow = (size_t)(n0 + lrow) * K;
  const int ldsOff = lrow * LDSW + lclh;
  const int aFrag0 = (wm*32 + rr) * LDSW + hh*8;   // + mt*16*LDSW
  const int bFrag0 = (wn*64 + rr) * LDSW + hh*8;   // + nt*16*LDSW

  // ---- prologue: stage K-tile 0 into LDS buffer 0 ----
  {
    u32x4 a0 = *(const u32x4*)(A  + aRow + lclh);
    u32x4 a1 = *(const u32x4*)(A  + aRow + lclh + 8);
    u32x4 b0 = *(const u32x4*)(Bt + bRow + lclh);
    u32x4 b1 = *(const u32x4*)(Bt + bRow + lclh + 8);
    *(u32x4*)&As[0][ldsOff]     = a0;
    *(u32x4*)&As[0][ldsOff + 8] = a1;
    *(u32x4*)&Bs[0][ldsOff]     = b0;
    *(u32x4*)&Bs[0][ldsOff + 8] = b1;
  }
  __syncthreads();

  const int nIter = K / BK;
  for (int it = 0; it < nIter; ++it) {
    const int cur = it & 1;
    const bool hasNext = (it + 1) < nIter;

    // Issue global loads for the next K-tile first; their latency is hidden
    // behind this tile's WMMA chain (consumed only at the ds_store below).
    u32x4 na0, na1, nb0, nb1;
    if (hasNext) {
      const int k1 = (it + 1) * BK;
      na0 = *(const u32x4*)(A  + aRow + k1 + lclh);
      na1 = *(const u32x4*)(A  + aRow + k1 + lclh + 8);
      nb0 = *(const u32x4*)(Bt + bRow + k1 + lclh);
      nb1 = *(const u32x4*)(Bt + bRow + k1 + lclh + 8);
      if (it + 2 < nIter) {                 // global_prefetch_b8 tile k+2
        __builtin_prefetch(A  + aRow + k1 + BK + lclh, 0, 0);
        __builtin_prefetch(Bt + bRow + k1 + BK + lclh, 0, 0);
      }
    }

    FragBF aF[2], bF[4];
    // A 16x32 fragment: lane-half h holds K chunks [8h,8h+8) and [16+8h,+8)
    #pragma unroll
    for (int mt = 0; mt < 2; ++mt) {
      int rb = aFrag0 + mt*16*LDSW;
      aF[mt].u[0] = *(const u32x4*)&As[cur][rb];
      aF[mt].u[1] = *(const u32x4*)&As[cur][rb + 16];
    }
    // B 32x16 fragment: lane-half h holds K chunks [8h,8h+8) and [16+8h,+8)
    #pragma unroll
    for (int nt = 0; nt < 4; ++nt) {
      int rb = bFrag0 + nt*16*LDSW;
      bF[nt].u[0] = *(const u32x4*)&Bs[cur][rb];
      bF[nt].u[1] = *(const u32x4*)&Bs[cur][rb + 16];
    }
    #pragma unroll
    for (int mt = 0; mt < 2; ++mt)
      #pragma unroll
      for (int nt = 0; nt < 4; ++nt)
        acc[mt][nt] = __builtin_amdgcn_wmma_f32_16x16x32_bf16(
            false, aF[mt].v, false, bF[nt].v,
            (short)0, acc[mt][nt], false, false);

    // Stage next tile into the other buffer; safe vs. this iteration's reads
    // (different buffer) and vs. last iteration's reads (previous barrier).
    if (hasNext) {
      const int nxt = cur ^ 1;
      *(u32x4*)&As[nxt][ldsOff]     = na0;
      *(u32x4*)&As[nxt][ldsOff + 8] = na1;
      *(u32x4*)&Bs[nxt][ldsOff]     = nb0;
      *(u32x4*)&Bs[nxt][ldsOff + 8] = nb1;
    }
    __syncthreads();
  }

  // C/D layout: VGPR v -> (M = v + 8*lane_half, N = lane&15)
  #pragma unroll
  for (int mt = 0; mt < 2; ++mt) {
    #pragma unroll
    for (int nt = 0; nt < 4; ++nt) {
      int row0 = m0 + wm*32 + mt*16 + hh*8;
      int col  = n0 + wn*64 + nt*16 + rr;
      if (EPI == 1) {
        unsigned short* C = (unsigned short*)Cv;
        #pragma unroll
        for (int v = 0; v < 8; ++v)
          C[(size_t)(row0 + v)*N + col] = f2bf(acc[mt][nt][v]);
      } else if (EPI == 2) {
        float* C = (float*)Cv;
        #pragma unroll
        for (int v = 0; v < 8; ++v) {
          size_t idx = (size_t)(row0 + v)*N + col;
          C[idx] = C[idx] + acc[mt][nt][v];
        }
      } else {
        float* C = (float*)Cv;
        #pragma unroll
        for (int v = 0; v < 8; ++v)
          C[(size_t)(row0 + v)*N + col] = acc[mt][nt][v];
      }
    }
  }
}

// ---------------------------------------------------------------------------
// rmsnorm(x, gamma) * sqrt(DIM), cast to bf16.  One block per row.
// ---------------------------------------------------------------------------
__global__ __launch_bounds__(256) void rmsnorm_cast_kernel(
    const float* __restrict__ x, const float* __restrict__ gamma,
    unsigned short* __restrict__ out)
{
  __shared__ float red[256];
  const int row = blockIdx.x;
  const int t = threadIdx.x;
  const float* xr = x + (size_t)row * DIM_;
  float s = 0.f;
  for (int i = t; i < DIM_; i += 256) { float v = xr[i]; s += v * v; }
  red[t] = s; __syncthreads();
  for (int off = 128; off > 0; off >>= 1) {
    if (t < off) red[t] += red[t + off];
    __syncthreads();
  }
  float inv = rsqrtf(red[0] + 1e-5f) * 27.712812921102035f;   // sqrt(768)
  for (int i = t; i < DIM_; i += 256)
    out[(size_t)row*DIM_ + i] = f2bf(xr[i] * inv * gamma[i]);
}

// ---------------------------------------------------------------------------
// fp32 (K,N) -> bf16 (N,K) transpose+cast via LDS tile.
// ---------------------------------------------------------------------------
__global__ __launch_bounds__(256) void transpose_cast_kernel(
    const float* __restrict__ in, unsigned short* __restrict__ out, int K, int N)
{
  __shared__ float tile[32][33];
  const int tx = threadIdx.x & 31;
  const int ty = threadIdx.x >> 5;
  const int kb = blockIdx.y * 32;
  const int nb = blockIdx.x * 32;
  #pragma unroll
  for (int s = 0; s < 32; s += 8) {
    int k = kb + ty + s, n = nb + tx;
    tile[ty + s][tx] = (k < K && n < N) ? in[(size_t)k*N + n] : 0.f;
  }
  __syncthreads();
  #pragma unroll
  for (int s = 0; s < 32; s += 8) {
    int n = nb + ty + s, k = kb + tx;
    if (n < N && k < K) out[(size_t)n*K + k] = f2bf(tile[tx][ty + s]);
  }
}

// ---------------------------------------------------------------------------
// tokens = [text_encodings | text_embed | time_emb_table[ts] | learned_query]
// ---------------------------------------------------------------------------
__global__ void build_tokens_kernel(
    const float* __restrict__ te, const float* __restrict__ tem,
    const float* __restrict__ tt, const float* __restrict__ lq,
    const int* __restrict__ ts, float* __restrict__ x)
{
  size_t idx = (size_t)blockIdx.x * blockDim.x + threadIdx.x;
  if (idx >= (size_t)B_ * NTOK * DIM_) return;
  int d   = (int)(idx % DIM_);
  int tok = (int)((idx / DIM_) % NTOK);
  int b   = (int)(idx / ((size_t)DIM_ * NTOK));
  float v;
  if (tok < L_)          v = te[((size_t)b*L_ + tok)*DIM_ + d];
  else if (tok == L_)    v = tem[(size_t)b*DIM_ + d];
  else if (tok == L_+1)  v = tt[(size_t)ts[b]*DIM_ + d];
  else                   v = lq[d];
  x[idx] = v;
}

// ---------------------------------------------------------------------------
// Relative position bias (HEADS, NTOK, NKEY) from (BUCKETS, HEADS) table.
// ---------------------------------------------------------------------------
__global__ void rel_bias_kernel(const float* __restrict__ table,
                                float* __restrict__ bias)
{
  int idx = blockIdx.x * blockDim.x + threadIdx.x;
  if (idx >= HEADS_ * NTOK * NKEY) return;
  int j = idx % NKEY;
  int i = (idx / NKEY) % NTOK;
  int h = idx / (NKEY * NTOK);
  int n = i - j; if (n < 0) n = 0;           // n = max(-(k - q), 0)
  int bucket;
  if (n < 16) bucket = n;
  else {
    int vl = 16 + (int)(logf((float)n * (1.f/16.f)) * (16.f / logf(8.f)));
    bucket = vl < 31 ? vl : 31;
  }
  bias[idx] = table[bucket * HEADS_ + h];
}

// ---------------------------------------------------------------------------
// k,v (B, 81, 64) fp32 with null_kv at position 0; kv comes from the
// Wkv GEMM as (B*80, 128) fp32.
// ---------------------------------------------------------------------------
__global__ void assemble_kv_kernel(const float* __restrict__ kv,
                                   const float* __restrict__ nullkv,
                                   float* __restrict__ Kf, float* __restrict__ Vf)
{
  int d = threadIdx.x;               // 64
  int j = blockIdx.x % NKEY;
  int b = blockIdx.x / NKEY;
  size_t o = ((size_t)b*NKEY + j)*DH_ + d;
  if (j == 0) { Kf[o] = nullkv[d]; Vf[o] = nullkv[DH_ + d]; }
  else {
    size_t s = ((size_t)b*NTOK + (j - 1)) * (2*DH_);
    Kf[o] = kv[s + d];
    Vf[o] = kv[s + DH_ + d];
  }
}

// ---------------------------------------------------------------------------
// Attention per (b, head): two-pass fp32 softmax with k/v/mask in LDS.
// Only ~1.5% of total FLOPs, so VALU fp32 keeps full reference semantics
// (null kv, padding mask incl. any(mask), causal j <= i+1, bias).
// ---------------------------------------------------------------------------
__global__ __launch_bounds__(128) void attention_kernel(
    const unsigned short* __restrict__ qb,      // (B,80,768) bf16
    const float* __restrict__ Kf, const float* __restrict__ Vf, // (B,81,64)
    const float* __restrict__ bias,             // (12,80,81)
    const unsigned char* __restrict__ mask,     // (B,77) bool
    unsigned short* __restrict__ outb)          // (B,80,768) bf16
{
  __shared__ float kS[NKEY * DH_];
  __shared__ float vS[NKEY * DH_];
  __shared__ unsigned char mS[NKEY];
  const int b  = blockIdx.x / HEADS_;
  const int hd = blockIdx.x % HEADS_;
  const int t  = threadIdx.x;

  for (int i = t; i < NKEY * DH_; i += 128) {
    kS[i] = Kf[(size_t)b*NKEY*DH_ + i];
    vS[i] = Vf[(size_t)b*NKEY*DH_ + i];
  }
  if (t < L_) mS[1 + t] = mask[(size_t)b*L_ + t] ? 1 : 0;
  if (t == 0) { mS[0] = 1; mS[L_+2] = 1; mS[L_+3] = 1; }
  __syncthreads();
  if (t == 0) {                       // any(mask) gates the text_embed token
    unsigned char any = 0;
    for (int j = 1; j <= L_; ++j) any |= mS[j];
    mS[L_+1] = any;
  }
  __syncthreads();

  const int i = t;
  if (i < NTOK) {
    float q[DH_];
    const unsigned short* qp = qb + ((size_t)b*NTOK + i)*INNER_ + hd*DH_;
    #pragma unroll
    for (int d = 0; d < DH_; ++d) q[d] = bf2f(qp[d]) * 0.125f;  // DH^-0.5

    const float* brow = bias + ((size_t)hd*NTOK + i)*NKEY;
    const int jmax = i + 1;           // causal: keys 0..i+1 visible

    float mx = -3.0e38f;
    for (int j = 0; j <= jmax; ++j) {
      if (!mS[j]) continue;
      float s = brow[j];
      const float* kr = &kS[j * DH_];
      #pragma unroll
      for (int d = 0; d < DH_; ++d) s += q[d] * kr[d];
      mx = fmaxf(mx, s);
    }

    float o[DH_];
    #pragma unroll
    for (int d = 0; d < DH_; ++d) o[d] = 0.f;
    float sum = 0.f;
    for (int j = 0; j <= jmax; ++j) {
      if (!mS[j]) continue;
      float s = brow[j];
      const float* kr = &kS[j * DH_];
      #pragma unroll
      for (int d = 0; d < DH_; ++d) s += q[d] * kr[d];
      float e = __expf(s - mx);
      sum += e;
      const float* vr = &vS[j * DH_];
      #pragma unroll
      for (int d = 0; d < DH_; ++d) o[d] += e * vr[d];
    }
    float rs = 1.f / sum;
    unsigned short* op = outb + ((size_t)b*NTOK + i)*INNER_ + hd*DH_;
    #pragma unroll
    for (int d = 0; d < DH_; ++d) op[d] = f2bf(o[d] * rs);
  }
}

// ---------------------------------------------------------------------------
// g = a * silu(gate) from h = [a | gate] (bf16, width 2*FFIN) -> bf16.
// ---------------------------------------------------------------------------
__global__ void silu_gate_kernel(const unsigned short* __restrict__ h,
                                 unsigned short* __restrict__ g)
{
  size_t idx = (size_t)blockIdx.x * blockDim.x + threadIdx.x;
  if (idx >= (size_t)M_ * FFIN) return;
  size_t m = idx / FFIN;
  int n = (int)(idx % FFIN);
  float a  = bf2f(h[m*(2*FFIN) + n]);
  float gt = bf2f(h[m*(2*FFIN) + FFIN + n]);
  g[idx] = f2bf(a * gt / (1.f + __expf(-gt)));
}

__global__ void last_token_kernel(const float* __restrict__ x,
                                  float* __restrict__ out)
{
  int idx = blockIdx.x * blockDim.x + threadIdx.x;
  if (idx >= B_ * DIM_) return;
  int b = idx / DIM_, d = idx % DIM_;
  out[idx] = x[((size_t)b*NTOK + (NTOK - 1))*DIM_ + d];
}

// ---------------------------------------------------------------------------
// Host orchestration.
// ---------------------------------------------------------------------------
extern "C" void kernel_launch(void* const* d_in, const int* in_sizes, int n_in,
                              void* d_out, int out_size, void* d_ws, size_t ws_size,
                              hipStream_t stream)
{
  (void)in_sizes; (void)n_in; (void)out_size; (void)ws_size;
  const float* text_enc   = (const float*)d_in[1];
  const float* text_embed = (const float*)d_in[2];
  const float* time_table = (const float*)d_in[3];
  const float* lquery     = (const float*)d_in[4];
  const float* rel_table  = (const float*)d_in[5];
  const float* attn_gamma = (const float*)d_in[6];
  const float* Wq         = (const float*)d_in[7];
  const float* Wkv        = (const float*)d_in[8];
  const float* Wout       = (const float*)d_in[9];
  const float* null_kv    = (const float*)d_in[10];
  const float* ff_gamma   = (const float*)d_in[11];
  const float* Wff1       = (const float*)d_in[12];
  const float* Wff2       = (const float*)d_in[13];
  const int*   tsteps     = (const int*)d_in[14];
  const unsigned char* mask = (const unsigned char*)d_in[15];

  char* ws = (char*)d_ws;
  size_t off = 0;
  auto alloc = [&](size_t bytes) -> void* {
    void* p = ws + off;
    off += (bytes + 255) & ~(size_t)255;
    return p;
  };

  unsigned short* WqT   = (unsigned short*)alloc((size_t)DEPTH_*INNER_*DIM_*2);
  unsigned short* WkvT  = (unsigned short*)alloc((size_t)DEPTH_*2*DH_*DIM_*2);
  unsigned short* WoutT = (unsigned short*)alloc((size_t)DEPTH_*DIM_*INNER_*2);
  unsigned short* Wff1T = (unsigned short*)alloc((size_t)DEPTH_*2*FFIN*DIM_*2);
  unsigned short* Wff2T = (unsigned short*)alloc((size_t)DEPTH_*DIM_*FFIN*2);
  float*          x     = (float*)alloc((size_t)M_*DIM_*4);
  unsigned short* xn    = (unsigned short*)alloc((size_t)M_*DIM_*2);
  unsigned short* qbuf  = (unsigned short*)alloc((size_t)M_*INNER_*2);
  float*          kvbuf = (float*)alloc((size_t)M_*2*DH_*4);
  float*          Kf    = (float*)alloc((size_t)B_*NKEY*DH_*4);
  float*          Vf    = (float*)alloc((size_t)B_*NKEY*DH_*4);
  unsigned short* aout  = (unsigned short*)alloc((size_t)M_*INNER_*2);
  unsigned short* hbuf  = (unsigned short*)alloc((size_t)M_*2*FFIN*2);
  unsigned short* gbuf  = (unsigned short*)alloc((size_t)M_*FFIN*2);
  float*          biasb = (float*)alloc((size_t)HEADS_*NTOK*NKEY*4);

  // ---- one-time weight transpose + cast (fp32 -> bf16, W -> W^T) ----
  for (int l = 0; l < DEPTH_; ++l) {
    transpose_cast_kernel<<<dim3(INNER_/32, DIM_/32), 256, 0, stream>>>(
        Wq + (size_t)l*DIM_*INNER_, WqT + (size_t)l*INNER_*DIM_, DIM_, INNER_);
    transpose_cast_kernel<<<dim3((2*DH_)/32, DIM_/32), 256, 0, stream>>>(
        Wkv + (size_t)l*DIM_*2*DH_, WkvT + (size_t)l*2*DH_*DIM_, DIM_, 2*DH_);
    transpose_cast_kernel<<<dim3(DIM_/32, INNER_/32), 256, 0, stream>>>(
        Wout + (size_t)l*INNER_*DIM_, WoutT + (size_t)l*DIM_*INNER_, INNER_, DIM_);
    transpose_cast_kernel<<<dim3((2*FFIN)/32, DIM_/32), 256, 0, stream>>>(
        Wff1 + (size_t)l*DIM_*2*FFIN, Wff1T + (size_t)l*2*FFIN*DIM_, DIM_, 2*FFIN);
    transpose_cast_kernel<<<dim3(DIM_/32, FFIN/32), 256, 0, stream>>>(
        Wff2 + (size_t)l*FFIN*DIM_, Wff2T + (size_t)l*DIM_*FFIN, FFIN, DIM_);
  }

  {
    size_t tot = (size_t)B_ * NTOK * DIM_;
    build_tokens_kernel<<<(unsigned)((tot + 255) / 256), 256, 0, stream>>>(
        text_enc, text_embed, time_table, lquery, tsteps, x);
    int btot = HEADS_ * NTOK * NKEY;
    rel_bias_kernel<<<(btot + 255) / 256, 256, 0, stream>>>(rel_table, biasb);
  }

  // ---- 12 transformer layers ----
  for (int l = 0; l < DEPTH_; ++l) {
    rmsnorm_cast_kernel<<<M_, 256, 0, stream>>>(x, attn_gamma + (size_t)l*DIM_, xn);
    gemm_bf16_wmma<1><<<dim3(INNER_/BN, M_/BM), 256, 0, stream>>>(
        xn, WqT + (size_t)l*INNER_*DIM_, qbuf, M_, INNER_, DIM_);
    gemm_bf16_wmma<0><<<dim3((2*DH_)/BN, M_/BM), 256, 0, stream>>>(
        xn, WkvT + (size_t)l*2*DH_*DIM_, kvbuf, M_, 2*DH_, DIM_);
    assemble_kv_kernel<<<B_*NKEY, DH_, 0, stream>>>(
        kvbuf, null_kv + (size_t)l*2*DH_, Kf, Vf);
    attention_kernel<<<B_*HEADS_, 128, 0, stream>>>(
        qbuf, Kf, Vf, biasb, mask, aout);
    gemm_bf16_wmma<2><<<dim3(DIM_/BN, M_/BM), 256, 0, stream>>>(   // x += out@Wout
        aout, WoutT + (size_t)l*DIM_*INNER_, x, M_, DIM_, INNER_);

    rmsnorm_cast_kernel<<<M_, 256, 0, stream>>>(x, ff_gamma + (size_t)l*DIM_, xn);
    gemm_bf16_wmma<1><<<dim3((2*FFIN)/BN, M_/BM), 256, 0, stream>>>(
        xn, Wff1T + (size_t)l*2*FFIN*DIM_, hbuf, M_, 2*FFIN, DIM_);
    size_t gt = (size_t)M_ * FFIN;
    silu_gate_kernel<<<(unsigned)((gt + 255) / 256), 256, 0, stream>>>(hbuf, gbuf);
    gemm_bf16_wmma<2><<<dim3(DIM_/BN, M_/BM), 256, 0, stream>>>(   // x += g@Wff2
        gbuf, Wff2T + (size_t)l*DIM_*FFIN, x, M_, DIM_, FFIN);
  }

  last_token_kernel<<<(B_*DIM_ + 255) / 256, 256, 0, stream>>>(x, (float*)d_out);
}
